// M2GNN_one_GAT_62277025792394
// MI455X (gfx1250) — matrix operand order repaired and checked by reading, not compiled, not run on hardware
//
#include <hip/hip_runtime.h>
#include <hip/hip_bf16.h>

// ---------------------------------------------------------------------------
// M2GNN forward for MI455X (gfx1250, wave32, WMMA).
// All dense projections run through v_wmma_f32_16x16x32_f16 (f16 in, f32 acc),
// with fragment-ordered LDS staging (wide ds_load_b128 fragment loads) and
// 4 WMMAs per wave per K-step. global_prefetch_b8 prefetches the next K tile.
// Irregular GAT edge-softmax uses encoded-uint atomicMax + atomicAdd_f32.
// ---------------------------------------------------------------------------

typedef __attribute__((ext_vector_type(16))) _Float16 v16h;
typedef __attribute__((ext_vector_type(8)))  float    v8f;

#define DEV __device__ __forceinline__

constexpr int Bn = 8192;      // users
constexpr int Un = 65536;     // reviews
constexpr int Tn = 16384;     // tags
constexpr int En = 262144;    // edges per edge type
constexpr int Dd = 128;       // dim
constexpr int Hh = 8;         // heads
// Dh = 16

// ---------------------------------------------------------------------------
// Generic WMMA GEMM:  C[M,N] = act(A[M,K] @ W[K,N] + bias)
// A,W f32 in global, converted to f16 into LDS *in fragment order*.
// Block: 256 threads = 8 waves; block tile 128x64; wave tile 32x32 (2x2 WMMA).
// Requires K % 32 == 0, N % 64 == 0.
//
// Fragment layouts (ISA 7.12.2, wave32):
//  A 16x32 f16 : lane l -> row (l&15); elems 0-7 -> K = ko+0..7,
//                elems 8-15 -> K = ko+16..23, ko = (l>=16 ? 8 : 0)
//  B 32x16 f16 : lane l -> col (l&15); elems 0-7 -> K = ko+0..7,
//                elems 8-15 -> K = ko+16..23 (same order as A)
//  C/D f32     : VGPR r -> M = r + (l>=16 ? 8 : 0), N = l&15
// ---------------------------------------------------------------------------
__global__ __launch_bounds__(256)
void k_gemm(const float* __restrict__ A, int ldA,
            const float* __restrict__ W, int ldW,
            float* __restrict__ C, int ldC,
            int M, int N, int K,
            const float* __restrict__ bias, int relu)
{
    // fragment-ordered staging: [subtile][lane][elem], 32B contiguous per lane
    __shared__ __align__(32) _Float16 Af[8][32][16];   // 8 KB : 128 rows x 32 K
    __shared__ __align__(32) _Float16 Bf[4][32][16];   // 4 KB : 32 K x 64 cols
    const int tid  = threadIdx.x;
    const int lane = tid & 31;
    const int wave = tid >> 5;
    const int wm   = (wave & 3) * 2;    // first of 2 M-subtiles for this wave
    const int wn   = (wave >> 2) * 2;   // first of 2 N-subtiles for this wave
    const int brow = blockIdx.y * 128;
    const int bcol = blockIdx.x * 64;

    v8f acc[2][2] = {};
    for (int kc = 0; kc < K; kc += 32) {
        // ---- A fill: 512 runs of 8 contiguous K values (r = t*64 + l*2 + half)
        for (int r = tid; r < 512; r += 256) {
            int half = r & 1;
            int l    = (r >> 1) & 31;
            int t    = r >> 6;
            int row  = brow + t * 16 + (l & 15);
            int k0   = ((l >> 4) << 3) + (half << 4);   // ko + (half ? 16 : 0)
            _Float16* dst = &Af[t][l][half * 8];
            if (row < M) {
                const float* g = &A[(size_t)row * ldA + kc + k0];
#pragma unroll
                for (int e = 0; e < 8; ++e) dst[e] = (_Float16)g[e];
            } else {
#pragma unroll
                for (int e = 0; e < 8; ++e) dst[e] = (_Float16)0.f;
            }
        }
        // ---- B fill: 256 runs of 8 K values strided by ldW (r = ns*64 + l*2 + half)
        for (int r = tid; r < 256; r += 256) {
            int half = r & 1;
            int l    = (r >> 1) & 31;
            int ns   = r >> 6;
            int col  = bcol + ns * 16 + (l & 15);
            int k0   = ((l >> 4) << 3) + (half << 4);   // ko + (half ? 16 : 0)
            const float* g = &W[(size_t)(kc + k0) * ldW + col];
            _Float16* dst = &Bf[ns][l][half * 8];
#pragma unroll
            for (int e = 0; e < 8; ++e) dst[e] = (_Float16)g[(size_t)e * ldW];
        }
        __syncthreads();

        // prefetch next A K-chunk into cache (gfx1250 global_prefetch_b8)
        if (kc + 32 < K) {
            int prow = brow + (tid >> 1);
            if (prow < M)
                __builtin_prefetch(&A[(size_t)prow * ldA + kc + 32 + (tid & 1) * 16], 0, 1);
        }

        // fragment loads: 2x ds_load_b128 per v16h
        v16h a0 = *(const v16h*)Af[wm + 0][lane];
        v16h a1 = *(const v16h*)Af[wm + 1][lane];
        v16h b0 = *(const v16h*)Bf[wn + 0][lane];
        v16h b1 = *(const v16h*)Bf[wn + 1][lane];
        acc[0][0] = __builtin_amdgcn_wmma_f32_16x16x32_f16(false, a0, false, b0,
                                                           (short)0, acc[0][0], false, false);
        acc[0][1] = __builtin_amdgcn_wmma_f32_16x16x32_f16(false, a0, false, b1,
                                                           (short)0, acc[0][1], false, false);
        acc[1][0] = __builtin_amdgcn_wmma_f32_16x16x32_f16(false, a1, false, b0,
                                                           (short)0, acc[1][0], false, false);
        acc[1][1] = __builtin_amdgcn_wmma_f32_16x16x32_f16(false, a1, false, b1,
                                                           (short)0, acc[1][1], false, false);
        __syncthreads();
    }

#pragma unroll
    for (int i = 0; i < 2; ++i)
#pragma unroll
        for (int j = 0; j < 2; ++j) {
            int gn  = bcol + (wn + j) * 16 + (lane & 15);
            float bv = bias ? bias[gn] : 0.f;
#pragma unroll
            for (int r = 0; r < 8; ++r) {
                int m = brow + (wm + i) * 16 + r + ((lane >> 4) << 3);
                if (m < M) {
                    float v = acc[i][j][r] + bv;
                    if (relu) v = v > 0.f ? v : 0.f;
                    C[(size_t)m * ldC + gn] = v;
                }
            }
        }
}

// ---------------------------------------------------------------------------
// helpers
// ---------------------------------------------------------------------------
DEV float lrelu02(float x) { return x > 0.f ? x : 0.2f * x; }
DEV unsigned enc_f(float f) {               // monotone float -> uint key
    unsigned u = __float_as_uint(f);
    return (u & 0x80000000u) ? ~u : (u | 0x80000000u);
}
DEV float wred(float v) {                   // wave32 all-reduce sum
    for (int o = 16; o; o >>= 1) v += __shfl_xor(v, o, 32);
    return v;
}

__global__ void k_fill_u32(unsigned* p, int n, unsigned v) {
    int i = blockIdx.x * 256 + threadIdx.x; if (i < n) p[i] = v;
}
__global__ void k_fill_f32(float* p, int n, float v) {
    int i = blockIdx.x * 256 + threadIdx.x; if (i < n) p[i] = v;
}

// ---------------------------------------------------------------------------
// low capsules
// ---------------------------------------------------------------------------
__global__ void k_emb(const float* __restrict__ u, const float* __restrict__ fW,
                      const float* __restrict__ fb, float* __restrict__ emb)
{
    int idx = blockIdx.x * 256 + threadIdx.x;
    if (idx >= Bn * 8 * 128) return;
    int d = idx & 127, n = (idx >> 7) & 7, b = idx >> 10;
    emb[idx] = u[b * 8 + n] * fW[n * 128 + d] + fb[n * 128 + d];
}

__global__ void k_gate(const float* __restrict__ h1, const float* __restrict__ g2W,
                       const float* __restrict__ g2b, float* __restrict__ gate)
{
    int idx = blockIdx.x * 256 + threadIdx.x;
    if (idx >= Bn * 8) return;
    int n = idx & 7;
    const float* hp = h1 + (size_t)idx * 128;
    float s = g2b[n];
    for (int d = 0; d < 128; ++d) s += hp[d] * g2W[n * 128 + d];
    gate[idx] = 1.f / (1.f + __expf(-s));
}

__global__ void k_hfeat(const float* __restrict__ gate, const float* __restrict__ emb,
                        float* __restrict__ hf)
{
    int idx = blockIdx.x * 256 + threadIdx.x;
    if (idx >= Bn * 128) return;
    int d = idx & 127, b = idx >> 7;
    float s = 0.f;
    for (int n = 0; n < 8; ++n) s += gate[b * 8 + n] * emb[(size_t)(b * 8 + n) * 128 + d];
    hf[idx] = s * 0.125f;
}

// ---------------------------------------------------------------------------
// GAT edge kernels
// ---------------------------------------------------------------------------
__global__ void k_el(const float* __restrict__ z, const float* __restrict__ av,
                     float* __restrict__ e, int NH)
{
    int idx = blockIdx.x * 256 + threadIdx.x;
    if (idx >= NH) return;
    int h = idx & 7, i = idx >> 3;
    const float* zp = z + (size_t)i * 128 + h * 16;
    const float* ap = av + h * 16;
    float s = 0.f;
#pragma unroll
    for (int d = 0; d < 16; ++d) s += zp[d] * ap[d];
    e[idx] = s;
}

__global__ void k_edge_max(const float* __restrict__ el, const float* __restrict__ er,
                           const int* __restrict__ src, const int* __restrict__ dst,
                           unsigned* __restrict__ men)
{
    int idx = blockIdx.x * 256 + threadIdx.x;
    if (idx >= En * Hh) return;
    int eidx = idx >> 3, h = idx & 7;
    int s = src[eidx], d = dst[eidx];
    float e = lrelu02(el[s * 8 + h] + er[d * 8 + h]);
    atomicMax(&men[d * 8 + h], enc_f(e));
}

__global__ void k_decode_m(unsigned* __restrict__ men, int n)
{
    int i = blockIdx.x * 256 + threadIdx.x;
    if (i >= n) return;
    unsigned u = men[i];
    unsigned raw = (u & 0x80000000u) ? (u & 0x7fffffffu) : ~u;
    float v = __uint_as_float(raw);
    if (!(v > -3.0e38f && v < 3.0e38f)) v = 0.f;  // isfinite -> else 0
    ((float*)men)[i] = v;
}

__global__ void k_edge_sum(const float* __restrict__ el, const float* __restrict__ er,
                           const float* __restrict__ m,
                           const int* __restrict__ src, const int* __restrict__ dst,
                           float* __restrict__ den)
{
    int idx = blockIdx.x * 256 + threadIdx.x;
    if (idx >= En * Hh) return;
    int eidx = idx >> 3, h = idx & 7;
    int s = src[eidx], d = dst[eidx];
    float e = lrelu02(el[s * 8 + h] + er[d * 8 + h]);
    atomicAdd(&den[d * 8 + h], __expf(e - m[d * 8 + h]));
}

__global__ void k_edge_scatter(const float* __restrict__ zs,
                               const float* __restrict__ el, const float* __restrict__ er,
                               const float* __restrict__ m, const float* __restrict__ den,
                               const int* __restrict__ src, const int* __restrict__ dst,
                               float* __restrict__ out)
{
    int idx = blockIdx.x * 256 + threadIdx.x;
    if (idx >= En * Hh) return;
    int eidx = idx >> 3, h = idx & 7;
    int s = src[eidx], d = dst[eidx];
    float e  = lrelu02(el[s * 8 + h] + er[d * 8 + h]);
    float ex = __expf(e - m[d * 8 + h]);
    float a  = ex / fmaxf(den[d * 8 + h], 1e-9f);
    const float* zrow = zs + (size_t)s * 128 + h * 16;
    float* orow = out + (size_t)d * 128 + h * 16;
#pragma unroll
    for (int k = 0; k < 16; ++k) atomicAdd(&orow[k], a * zrow[k]);
}

__global__ void k_set_bias_rows(float* __restrict__ out, int rows,
                                const float* __restrict__ b1,
                                const float* __restrict__ b2,
                                const float* __restrict__ b3)
{
    int idx = blockIdx.x * 256 + threadIdx.x;
    if (idx >= rows * 128) return;
    int d = idx & 127;
    float v = b1[d];
    if (b2) v += b2[d];
    if (b3) v += b3[d];
    out[idx] = v;
}

// ---------------------------------------------------------------------------
// routing 1: u_in is a broadcast of h_out -> u_hat independent of n.
// Collapses to scalar recurrences on q_m = ||uh[b,m,:]||^2.
// ---------------------------------------------------------------------------
__global__ __launch_bounds__(32)
void k_routing1(const float* __restrict__ uh, float* __restrict__ v1)
{
    int b = blockIdx.x, lane = threadIdx.x;
    const float* base = uh + (size_t)b * 512;
    float u[4][4], q[4];
    for (int m = 0; m < 4; ++m) {
        float p = 0.f;
#pragma unroll
        for (int j = 0; j < 4; ++j) {
            u[m][j] = base[m * 128 + lane + 32 * j];
            p += u[m][j] * u[m][j];
        }
        q[m] = wred(p);
    }
    float bb[4] = {0, 0, 0, 0}, sc[4];
    for (int it = 0; it < 3; ++it) {
        float mx = fmaxf(fmaxf(bb[0], bb[1]), fmaxf(bb[2], bb[3]));
        float ex[4], ssum = 0.f;
        for (int m = 0; m < 4; ++m) { ex[m] = __expf(bb[m] - mx); ssum += ex[m]; }
        float inv8 = 8.f / ssum;                 // sum over the 8 identical caps
        for (int m = 0; m < 4; ++m) {
            float c8 = ex[m] * inv8;             // s_m = c8 * uh_m
            float sn = c8 * c8 * q[m];
            float f  = sn / (1.f + sn) / (sqrtf(sn) + 1e-8f);
            sc[m] = f * c8;                      // v_m = sc * uh_m
            if (it < 2) bb[m] += sc[m] * q[m];   // <uh_m, v_m>
        }
    }
    for (int m = 0; m < 4; ++m)
#pragma unroll
        for (int j = 0; j < 4; ++j)
            v1[(size_t)b * 512 + m * 128 + lane + 32 * j] = sc[m] * u[m][j];
}

// ---------------------------------------------------------------------------
// mid-to-high GAT: dense K4 (no self loops) attention on z[b,4,H,128]
// ---------------------------------------------------------------------------
__global__ void k_bmean(const float* __restrict__ gb, float* __restrict__ bm)
{
    int d = threadIdx.x;
    float s = 0.f;
    for (int h = 0; h < 8; ++h) s += gb[h * 128 + d];
    bm[d] = s * 0.125f;
}

__global__ __launch_bounds__(128)
void k_gmh_attn(const float* __restrict__ zg, const float* __restrict__ al,
                const float* __restrict__ ar, const float* __restrict__ bmean,
                float* __restrict__ v1g)
{
    __shared__ float z[4][1024];
    __shared__ float el[4][8], er[4][8];
    __shared__ float alpha[4][4][8];
    int b = blockIdx.x, tid = threadIdx.x;
    const float* src = zg + (size_t)b * 4096;
    for (int t = tid; t < 4096; t += 128) z[t >> 10][t & 1023] = src[t];
    __syncthreads();
    if (tid < 32) {                               // el/er per (cap, head)
        int i = tid >> 3, h = tid & 7;
        float sl = 0.f, sr = 0.f;
        for (int d = 0; d < 128; ++d) {
            float zz = z[i][h * 128 + d];
            sl += zz * al[h * 128 + d];
            sr += zz * ar[h * 128 + d];
        }
        el[i][h] = sl; er[i][h] = sr;
    }
    __syncthreads();
    if (tid < 32) {                               // softmax over sources s != i
        int i = tid >> 3, h = tid & 7;
        float e[4], mx = -3.0e38f;
        for (int s = 0; s < 4; ++s) {
            if (s == i) { e[s] = -3.0e38f; continue; }
            float x = lrelu02(el[s][h] + er[i][h]);
            e[s] = x; if (x > mx) mx = x;
        }
        float dn = 0.f;
        for (int s = 0; s < 4; ++s) {
            float ex = (s == i) ? 0.f : __expf(e[s] - mx);
            alpha[i][s][h] = ex; dn += ex;
        }
        float inv = 1.f / dn;
        for (int s = 0; s < 4; ++s) alpha[i][s][h] *= inv;
    }
    __syncthreads();
    for (int t = tid; t < 512; t += 128) {        // head-avg output + bias mean
        int i = t >> 7, d = t & 127;
        float o = 0.f;
        for (int h = 0; h < 8; ++h) {
            float a0 = 0.f;
            for (int s = 0; s < 4; ++s) a0 += alpha[i][s][h] * z[s][h * 128 + d];
            o += a0;
        }
        v1g[((size_t)b * 4 + i) * 128 + d] = o * 0.125f + bmean[d];
    }
}

// ---------------------------------------------------------------------------
// routing 2: full dynamic routing on uhat[b,4,2,128]
// ---------------------------------------------------------------------------
__global__ __launch_bounds__(32)
void k_routing2(const float* __restrict__ uhat, float* __restrict__ v2)
{
    int b = blockIdx.x, lane = threadIdx.x;
    const float* base = uhat + (size_t)b * 1024;
    float uu[4][2][4];
    for (int n = 0; n < 4; ++n)
        for (int m = 0; m < 2; ++m)
#pragma unroll
            for (int j = 0; j < 4; ++j)
                uu[n][m][j] = base[(n * 2 + m) * 128 + lane + 32 * j];
    float bb[4][2] = {}, vv[2][4];
    for (int it = 0; it < 3; ++it) {
        float c[4][2];
        for (int n = 0; n < 4; ++n) {
            float mx = fmaxf(bb[n][0], bb[n][1]);
            float e0 = __expf(bb[n][0] - mx), e1 = __expf(bb[n][1] - mx);
            float inv = 1.f / (e0 + e1);
            c[n][0] = e0 * inv; c[n][1] = e1 * inv;
        }
        for (int m = 0; m < 2; ++m) {
            float s[4];
#pragma unroll
            for (int j = 0; j < 4; ++j) {
                s[j] = 0.f;
                for (int n = 0; n < 4; ++n) s[j] += c[n][m] * uu[n][m][j];
            }
            float p = 0.f;
#pragma unroll
            for (int j = 0; j < 4; ++j) p += s[j] * s[j];
            p = wred(p);
            float f = p / (1.f + p) / (sqrtf(p) + 1e-8f);
#pragma unroll
            for (int j = 0; j < 4; ++j) vv[m][j] = f * s[j];
        }
        if (it < 2)
            for (int n = 0; n < 4; ++n)
                for (int m = 0; m < 2; ++m) {
                    float p = 0.f;
#pragma unroll
                    for (int j = 0; j < 4; ++j) p += uu[n][m][j] * vv[m][j];
                    bb[n][m] += wred(p);
                }
    }
    for (int m = 0; m < 2; ++m)
#pragma unroll
        for (int j = 0; j < 4; ++j)
            v2[(size_t)b * 256 + m * 128 + lane + 32 * j] = vv[m][j];
}

// ---------------------------------------------------------------------------
// gating head
// ---------------------------------------------------------------------------
__global__ void k_act(const float* __restrict__ gN1, const float* __restrict__ W2,
                      const float* __restrict__ b2, float* __restrict__ act)
{
    int b = blockIdx.x * 256 + threadIdx.x;
    if (b >= Bn) return;
    float s0 = b2[0], s1 = b2[1];
    const float* g = gN1 + (size_t)b * 128;
    for (int d = 0; d < 128; ++d) { s0 += g[d] * W2[d * 2]; s1 += g[d] * W2[d * 2 + 1]; }
    act[b * 2 + 0] = 1.f / (1.f + __expf(-s0));
    act[b * 2 + 1] = 1.f / (1.f + __expf(-s1));
}

__global__ void k_caps(const float* __restrict__ v2, const float* __restrict__ act,
                       float* __restrict__ caps)
{
    int idx = blockIdx.x * 256 + threadIdx.x;
    if (idx >= Bn * 256) return;
    int b = idx >> 8, j = (idx >> 7) & 1;
    caps[idx] = v2[idx] * act[b * 2 + j];
}

// ---------------------------------------------------------------------------
// host launcher
// ---------------------------------------------------------------------------
extern "C" void kernel_launch(void* const* d_in, const int* in_sizes, int n_in,
                              void* d_out, int out_size, void* d_ws, size_t ws_size,
                              hipStream_t stream)
{
    (void)in_sizes; (void)n_in; (void)out_size; (void)ws_size;
    // ---- inputs (setup_inputs dict insertion order; params dict expanded in
    //      insertion order; each GAT dict is W, al, ar, b) ----
    const float* user_embed = (const float*)d_in[0];   // [B,8]
    const float* review     = (const float*)d_in[1];   // [U,128]
    const float* tag        = (const float*)d_in[2];   // [T,128]
    const int*   e_int      = (const int*)d_in[3];
    const int*   e_hkt      = (const int*)d_in[4];
    const int*   e_ukt      = (const int*)d_in[5];
    const int*   e_t2t      = (const int*)d_in[6];
    const int*   e_hukt     = (const int*)d_in[7];
    const int*   e_test     = (const int*)d_in[8];
    const float* feat_W  = (const float*)d_in[9];
    const float* feat_b  = (const float*)d_in[10];
    const float* g1W     = (const float*)d_in[11];
    const float* g1b     = (const float*)d_in[12];
    const float* g2W     = (const float*)d_in[13];
    const float* g2b     = (const float*)d_in[14];
    const float* gat_p[6][4];                      // int, hkt, ukt, t2t, hukt, test
    for (int g = 0; g < 6; ++g)
        for (int k = 0; k < 4; ++k) gat_p[g][k] = (const float*)d_in[15 + g * 4 + k];
    const float* gmhW  = (const float*)d_in[39];   // [128,1024]
    const float* gmhal = (const float*)d_in[40];   // [8,128]
    const float* gmhar = (const float*)d_in[41];
    const float* gmhb  = (const float*)d_in[42];   // [1024]
    const float* cw1   = (const float*)d_in[43];   // [4,128,128]
    const float* cw2   = (const float*)d_in[44];   // [2,128,128]
    const float* gNW1  = (const float*)d_in[45];
    const float* gNb1  = (const float*)d_in[46];
    const float* gNW2  = (const float*)d_in[47];   // [128,2]
    const float* gNb2  = (const float*)d_in[48];
    const float* projW = (const float*)d_in[49];   // [256,128]
    const float* projb = (const float*)d_in[50];

    float* out0  = (float*)d_out;                       // [B,128]
    float* u_out = out0 + (size_t)Bn * 128;             // [U,128]
    float* t_out = u_out + (size_t)Un * 128;            // [T,128]

    // ---- workspace layout (floats), heavy buffers time-aliased ----
    float* f = (float*)d_ws;
    float*    hfeat = f;                                // 1,048,576
    float*    hout  = hfeat + (size_t)Bn * 128;         // 1,048,576
    float*    zs    = hout  + (size_t)Bn * 128;         // 8,388,608 (alias: emb, uhat2)
    float*    zd    = zs    + (size_t)Un * 128;         // 8,388,608 (alias: h1, v2/gN1/caps)
    float*    el    = zd    + (size_t)Un * 128;         // 524,288 (alias: gate)
    float*    er    = el    + (size_t)Un * 8;           // 524,288
    unsigned* men   = (unsigned*)(er + (size_t)Un * 8); // 524,288
    float*    den   = (float*)men + (size_t)Un * 8;     // 524,288 (alias: act)
    float*    uh    = den   + (size_t)Un * 8;           // 4,194,304
    float*    v1    = uh    + (size_t)Bn * 512;         // 4,194,304
    float*    zg    = v1    + (size_t)Bn * 512;         // 33,554,432
    float*    v1g   = zg    + (size_t)Bn * 4096;        // 4,194,304
    float*    bmean = v1g   + (size_t)Bn * 512;         // 128
    float*    emb   = zs;                               // [B,8,128]
    float*    h1    = zd;                               // [B,8,128]
    float*    gate  = el;                               // [B,8]
    float*    uhat2 = zs;                               // [B,4,2,128]
    float*    v2b   = zd;                               // [B,2,128]
    float*    gN1   = v2b + (size_t)Bn * 256;           // [B,128]
    float*    caps  = gN1 + (size_t)Bn * 128;           // [B,256]
    float*    actb  = den;                              // [B,2]

    auto cdiv = [](int a, int b) { return (a + b - 1) / b; };
    auto gemm = [&](const float* A, int ldA, const float* W, int ldW,
                    float* C, int ldC, int M, int N, int K,
                    const float* bias, int relu) {
        dim3 grid(N / 64, cdiv(M, 128));
        k_gemm<<<grid, 256, 0, stream>>>(A, ldA, W, ldW, C, ldC, M, N, K, bias, relu);
    };

    // ===== low capsules =====
    k_emb<<<cdiv(Bn * 1024, 256), 256, 0, stream>>>(user_embed, feat_W, feat_b, emb);
    for (int n = 0; n < 8; ++n)
        gemm(emb + n * 128, 1024, g1W + (size_t)n * 16384, 128,
             h1 + n * 128, 1024, Bn, 128, 128, g1b + n * 128, /*relu=*/1);
    k_gate<<<cdiv(Bn * 8, 256), 256, 0, stream>>>(h1, g2W, g2b, gate);
    k_hfeat<<<cdiv(Bn * 128, 256), 256, 0, stream>>>(gate, emb, hfeat);

    // ===== hetero GATs =====
    auto run_gat = [&](const float* xs, int Ns, const float* xd, int Nd,
                       const float* W, const float* al, const float* ar,
                       const int* edges, float* outp, bool same) {
        gemm(xs, 128, W, 128, zs, 128, Ns, 128, 128, nullptr, 0);
        const float* zdp = zs;
        if (!same) { gemm(xd, 128, W, 128, zd, 128, Nd, 128, 128, nullptr, 0); zdp = zd; }
        k_el<<<cdiv(Ns * 8, 256), 256, 0, stream>>>(zs, al, el, Ns * 8);
        k_el<<<cdiv(Nd * 8, 256), 256, 0, stream>>>(zdp, ar, er, Nd * 8);
        k_fill_u32<<<cdiv(Nd * 8, 256), 256, 0, stream>>>(men, Nd * 8, 0u); // enc(-inf) floor
        const int *src = edges, *dst = edges + En;
        k_edge_max<<<cdiv(En * 8, 256), 256, 0, stream>>>(el, er, src, dst, men);
        k_decode_m<<<cdiv(Nd * 8, 256), 256, 0, stream>>>(men, Nd * 8);
        k_fill_f32<<<cdiv(Nd * 8, 256), 256, 0, stream>>>(den, Nd * 8, 0.f);
        k_edge_sum<<<cdiv(En * 8, 256), 256, 0, stream>>>(el, er, (float*)men, src, dst, den);
        k_edge_scatter<<<cdiv(En * 8, 256), 256, 0, stream>>>(zs, el, er, (float*)men, den,
                                                              src, dst, outp);
    };

    // h_out = GAT_int(review->user) + GAT_hkt(tag->user)   (biases pre-seeded)
    k_set_bias_rows<<<cdiv(Bn * 128, 256), 256, 0, stream>>>(hout, Bn, gat_p[0][3], gat_p[1][3], nullptr);
    run_gat(review, Un, hfeat, Bn, gat_p[0][0], gat_p[0][1], gat_p[0][2], e_int, hout, false);
    run_gat(tag,    Tn, hfeat, Bn, gat_p[1][0], gat_p[1][1], gat_p[1][2], e_hkt, hout, false);
    // u_out = GAT_ukt(user->review)
    k_set_bias_rows<<<cdiv(Un * 128, 256), 256, 0, stream>>>(u_out, Un, gat_p[2][3], nullptr, nullptr);
    run_gat(hfeat, Bn, review, Un, gat_p[2][0], gat_p[2][1], gat_p[2][2], e_ukt, u_out, false);
    // t_out = GAT_t2t + GAT_hukt + GAT_test
    k_set_bias_rows<<<cdiv(Tn * 128, 256), 256, 0, stream>>>(t_out, Tn, gat_p[3][3], gat_p[4][3], gat_p[5][3]);
    run_gat(tag,    Tn, tag, Tn, gat_p[3][0], gat_p[3][1], gat_p[3][2], e_t2t,  t_out, true);
    run_gat(hfeat,  Bn, tag, Tn, gat_p[4][0], gat_p[4][1], gat_p[4][2], e_hukt, t_out, false);
    run_gat(review, Un, tag, Tn, gat_p[5][0], gat_p[5][1], gat_p[5][2], e_test, t_out, false);

    // ===== routing low->mid (collapsed: uh[b,m,:] = hout[b] @ cw1[m]) =====
    for (int m = 0; m < 4; ++m)
        gemm(hout, 128, cw1 + (size_t)m * 16384, 128, uh + m * 128, 512, Bn, 128, 128, nullptr, 0);
    k_routing1<<<Bn, 32, 0, stream>>>(uh, v1);

    // ===== mid-to-high GAT =====
    gemm(v1, 128, gmhW, 1024, zg, 1024, Bn * 4, 1024, 128, nullptr, 0);
    k_bmean<<<1, 128, 0, stream>>>(gmhb, bmean);
    k_gmh_attn<<<Bn, 128, 0, stream>>>(zg, gmhal, gmhar, bmean, v1g);

    // ===== routing mid->high =====
    for (int m = 0; m < 2; ++m)
        gemm(v1g, 128, cw2 + (size_t)m * 16384, 128, uhat2 + m * 128, 256, Bn * 4, 128, 128, nullptr, 0);
    k_routing2<<<Bn, 32, 0, stream>>>(uhat2, v2b);

    // ===== gating + projection =====
    gemm(hout, 128, gNW1, 128, gN1, 128, Bn, 128, 128, gNb1, /*relu=*/1);
    k_act<<<cdiv(Bn, 256), 256, 0, stream>>>(gN1, gNW2, gNb2, actb);
    k_caps<<<cdiv(Bn * 256, 256), 256, 0, stream>>>(v2b, actb, caps);
    gemm(caps, 256, projW, 128, out0, 128, Bn, 128, 256, projb, 0);
}
